// MultilayerGRU_65214783422754
// MI455X (gfx1250) — matrix operand order repaired and checked by reading, not compiled, not run on hardware
//
#include <hip/hip_runtime.h>
#include <hip/hip_bf16.h>

// Problem constants (match the reference).
#define GB 128   // batch
#define GS 512   // seq len
#define GI 128   // input dim
#define GH 1024  // hidden dim
#define GO 128   // output dim
#define GL 2     // layers

typedef __attribute__((ext_vector_type(16))) __bf16 v16bf;
typedef __attribute__((ext_vector_type(8)))  float  v8f;

// ---------------------------------------------------------------------------
// CDNA5 async global->LDS copy (16B per lane), tracked by ASYNCcnt.
// LDS address is taken from the vdst VGPR; low 32 bits of the generic shared pointer
// are the wave-relative LDS offset.
// ---------------------------------------------------------------------------
__device__ __forceinline__ void async_copy_b128(void* lds_dst, const void* gsrc) {
    unsigned off = (unsigned)(uintptr_t)lds_dst;
    asm volatile("global_load_async_to_lds_b128 %0, %1, off"
                 :: "v"(off), "v"(gsrc)
                 : "memory");
}
__device__ __forceinline__ void wait_async_le1() {
    asm volatile("s_wait_asynccnt 0x1" ::: "memory");
}
__device__ __forceinline__ void wait_async_0() {
    asm volatile("s_wait_asynccnt 0x0" ::: "memory");
}

// ---------------------------------------------------------------------------
// WMMA fragment loaders (bf16 16x16x32, wave32).
// A-matrix 16x32 per ISA table: lane half=0 holds K=[k0..k0+7] then [k0+16..k0+23],
// half=1 holds K=[k0+8..k0+15] then [k0+24..k0+31] -> two contiguous 16B loads.
// B-matrix 32x16: lane half h holds K=[k0+8h..k0+8h+7] then [k0+16+8h..k0+23+8h] for
// column lane%16. B tiles are staged in LDS as [64 cols][32 K] contiguous.
// ---------------------------------------------------------------------------
__device__ __forceinline__ v16bf load_frag_a(const __bf16* A, int lda, int row,
                                             int k0, int half) {
    union { v16bf v; uint4 q[2]; } u;
    const __bf16* p = A + (long)row * lda + k0 + half * 8;
    u.q[0] = *reinterpret_cast<const uint4*>(p);
    u.q[1] = *reinterpret_cast<const uint4*>(p + 16);
    return u.v;
}

__device__ __forceinline__ v16bf load_frag_b_lds(const __bf16* sbuf, int col,
                                                 int half) {
    union { v16bf v; uint4 q[2]; } u;
    const uint4* p = reinterpret_cast<const uint4*>(sbuf + col * 32 + half * 8);
    u.q[0] = p[0];
    u.q[1] = p[2];
    return u.v;
}

// ---------------------------------------------------------------------------
// One K-segment of the GEMM: acc += A[m,:K] @ W[n,:K]^T for the workgroup's
// 128x64 tile. B (weights) double-buffered through LDS via async copies; A
// prefetched one chunk ahead into registers. Steady-state loop is branch-free;
// the final chunk is peeled so only it drains ASYNCcnt to 0.
// ---------------------------------------------------------------------------
__device__ __forceinline__ void gemm_segment(
    v8f acc[4], const __bf16* __restrict__ A, int lda,
    const __bf16* __restrict__ W, int K,
    int m0, int n0, int lane16, int half, int tid, __bf16* sB /* [2][2048] */) {
    const int nch   = K >> 5;          // 32-wide K chunks (>= 4 here)
    const int scol  = tid >> 2;        // 0..63: column this thread stages
    const int skoff = (tid & 3) * 8;   // 0,8,16,24: K offset (8 bf16 = 16B)
    const __bf16* wsrc = W + (long)(n0 + scol) * K + skoff;
    __bf16* sdst = sB + scol * 32 + skoff;
    const int arow = m0 + lane16;

    async_copy_b128(sdst, wsrc);                       // chunk 0 -> buf 0
    v16bf aNext = load_frag_a(A, lda, arow, 0, half);

    // Steady state: always one async copy in flight.
    for (int kc = 0; kc < nch - 1; ++kc) {
        const int b = kc & 1;
        async_copy_b128(sdst + (1 - b) * 2048, wsrc + (kc + 1) * 32);
        wait_async_le1();                              // chunk kc landed in LDS
        __syncthreads();

        v16bf a = aNext;
        aNext = load_frag_a(A, lda, arow, (kc + 1) * 32, half);

        const __bf16* sbuf = sB + b * 2048;
#pragma unroll
        for (int j = 0; j < 4; ++j) {
            v16bf bf = load_frag_b_lds(sbuf, j * 16 + lane16, half);
            acc[j] = __builtin_amdgcn_wmma_f32_16x16x32_bf16(
                false, a, false, bf, (short)0, acc[j], false, false);
        }
        __syncthreads();                               // buf b reusable next+1
    }

    // Tail chunk: drain and compute.
    wait_async_0();
    __syncthreads();
    const __bf16* sbuf = sB + ((nch - 1) & 1) * 2048;
#pragma unroll
    for (int j = 0; j < 4; ++j) {
        v16bf bf = load_frag_b_lds(sbuf, j * 16 + lane16, half);
        acc[j] = __builtin_amdgcn_wmma_f32_16x16x32_bf16(
            false, aNext, false, bf, (short)0, acc[j], false, false);
    }
    __syncthreads();                                   // sB reusable by caller
}

// ---------------------------------------------------------------------------
// Fused GEMM (+ optional second K-segment) + epilogue.
//   acc[m][n] = A1[m,:K1] @ W1[n,:K1]^T (+ A2[m,:K2] @ W2[n,:K2]^T) + bias[n]
// mode 0 (GRU update): g = tanh(acc); h = h*z + (1-z)*g; write h (f32) + hbf.
// mode 1 (plain):      out[m*outRowStride + n] = acc.
// Block = 256 threads = 8 waves; wave w owns M rows [16w,16w+16) (M must be 128);
// each wave computes 4 N-tiles (N chunk of 64 per workgroup, grid.x = N/64).
// ---------------------------------------------------------------------------
__global__ __launch_bounds__(256) void gemm_wmma_kernel(
    const __bf16* __restrict__ A1, int lda1, int K1, const __bf16* __restrict__ W1,
    const __bf16* __restrict__ A2, int lda2, int K2, const __bf16* __restrict__ W2,
    const float* __restrict__ bias, int N, int mode,
    const float* __restrict__ zbuf, float* __restrict__ hState,
    __bf16* __restrict__ hbf,
    float* __restrict__ out, long outRowStride) {
    __shared__ alignas(16) __bf16 sB[2][2048];  // 2 x 4KB double buffer

    const int tid    = threadIdx.x;
    const int wave   = tid >> 5;
    const int lane   = tid & 31;
    const int half   = lane >> 4;
    const int lane16 = lane & 15;
    const int m0     = wave * 16;
    const int n0     = blockIdx.x * 64;

    v8f acc[4];
#pragma unroll
    for (int j = 0; j < 4; ++j) acc[j] = v8f{};

    gemm_segment(acc, A1, lda1, W1, K1, m0, n0, lane16, half, tid, &sB[0][0]);
    if (A2 != nullptr)  // uniform branch: EXEC stays all-ones around WMMA
        gemm_segment(acc, A2, lda2, W2, K2, m0, n0, lane16, half, tid, &sB[0][0]);

    // C layout: acc[j][v] -> row m = m0 + v + 8*half, col n = n0 + j*16 + lane16.
#pragma unroll
    for (int j = 0; j < 4; ++j) {
        const int n  = n0 + j * 16 + lane16;
        const float bn = bias[n];
#pragma unroll
        for (int v = 0; v < 8; ++v) {
            const int m   = m0 + v + 8 * half;
            const float s = acc[j][v] + bn;
            if (mode == 0) {
                const float g  = tanhf(s);
                const float z  = zbuf[m];
                const long  ix = (long)m * N + n;
                const float hn = hState[ix] * z + (1.0f - z) * g;
                hState[ix] = hn;
                hbf[ix]    = (__bf16)hn;
            } else {
                out[(long)m * outRowStride + n] = s;
            }
        }
    }
}

// ---------------------------------------------------------------------------
// Scalar z/r gates: one 256-thread block per batch row.
//   z = sigmoid(inp.Wzx + bz + h.Wzh); r = sigmoid(inp.Wrx + br + h.Wrh)
// Writes zbuf[b] and rhbf[b,:] = bf16(r * h[b,:]).
// ---------------------------------------------------------------------------
__global__ __launch_bounds__(256) void gru_gates_kernel(
    const float* __restrict__ inp, long inpRowStride, int dIn,
    const float* __restrict__ h,
    const float* __restrict__ Wzx, const float* __restrict__ bz,
    const float* __restrict__ Wzh,
    const float* __restrict__ Wrx, const float* __restrict__ br,
    const float* __restrict__ Wrh,
    float* __restrict__ zbuf, __bf16* __restrict__ rhbf) {
    const int b   = blockIdx.x;
    const int tid = threadIdx.x;
    const float* xrow = inp + (long)b * inpRowStride;
    const float* hrow = h + (long)b * GH;

    float sz = 0.f, sr = 0.f;
    for (int i = tid; i < dIn; i += 256) {
        const float xv = xrow[i];
        sz += xv * Wzx[i];
        sr += xv * Wrx[i];
    }
    for (int j = tid; j < GH; j += 256) {
        const float hv = hrow[j];
        sz += hv * Wzh[j];
        sr += hv * Wrh[j];
    }

    __shared__ float red[2][256];
    __shared__ float zr[2];
    red[0][tid] = sz; red[1][tid] = sr;
    __syncthreads();
    for (int off = 128; off > 0; off >>= 1) {
        if (tid < off) {
            red[0][tid] += red[0][tid + off];
            red[1][tid] += red[1][tid + off];
        }
        __syncthreads();
    }
    if (tid == 0) {
        const float z = 1.f / (1.f + expf(-(red[0][0] + bz[0])));
        const float r = 1.f / (1.f + expf(-(red[1][0] + br[0])));
        zr[0] = z; zr[1] = r;
        zbuf[b] = z;
    }
    __syncthreads();
    const float r = zr[1];
    for (int j = tid; j < GH; j += 256)
        rhbf[(long)b * GH + j] = (__bf16)(r * hrow[j]);
}

// ---------------------------------------------------------------------------
// Utility kernels.
// ---------------------------------------------------------------------------
__global__ void f32_to_bf16_kernel(const float* __restrict__ in,
                                   __bf16* __restrict__ out, long n) {
    long i = (long)blockIdx.x * blockDim.x + threadIdx.x;
    const long stride = (long)gridDim.x * blockDim.x;
    for (; i < n; i += stride) out[i] = (__bf16)in[i];
}

__global__ void zero_h_kernel(float* __restrict__ h, __bf16* __restrict__ hbf,
                              long n) {
    long i = (long)blockIdx.x * blockDim.x + threadIdx.x;
    if (i < n) { h[i] = 0.f; hbf[i] = (__bf16)0.f; }
}

// h: [L][B][H] -> out: [B][L][H]
__global__ void copy_hidden_kernel(const float* __restrict__ h,
                                   float* __restrict__ out) {
    long i = (long)blockIdx.x * blockDim.x + threadIdx.x;
    const long n = (long)GL * GB * GH;
    if (i < n) {
        const int j = (int)(i % GH);
        const int b = (int)((i / GH) % GB);
        const int k = (int)(i / ((long)GH * GB));
        out[(long)b * GL * GH + (long)k * GH + j] = h[((long)k * GB + b) * GH + j];
    }
}

// ---------------------------------------------------------------------------
// Host launcher.
// d_in order: x, then per layer {Wzx,bz,Wzh,Wrx,br,Wrh,Wgx,bg,Wgh}, then Wo, bo.
// d_out: [B*S*O] layer_output floats then [B*L*H] hidden_state floats.
// ---------------------------------------------------------------------------
extern "C" void kernel_launch(void* const* d_in, const int* in_sizes, int n_in,
                              void* d_out, int out_size, void* d_ws, size_t ws_size,
                              hipStream_t stream) {
    (void)in_sizes; (void)n_in; (void)out_size; (void)ws_size;

    const float* x = (const float*)d_in[0];
    struct LayerP {
        const float *Wzx, *bz, *Wzh, *Wrx, *br, *Wrh, *Wgx, *bg, *Wgh;
    } lp[GL];
    int idx = 1;
    for (int k = 0; k < GL; ++k) {
        lp[k].Wzx = (const float*)d_in[idx++];
        lp[k].bz  = (const float*)d_in[idx++];
        lp[k].Wzh = (const float*)d_in[idx++];
        lp[k].Wrx = (const float*)d_in[idx++];
        lp[k].br  = (const float*)d_in[idx++];
        lp[k].Wrh = (const float*)d_in[idx++];
        lp[k].Wgx = (const float*)d_in[idx++];
        lp[k].bg  = (const float*)d_in[idx++];
        lp[k].Wgh = (const float*)d_in[idx++];
    }
    const float* Wo = (const float*)d_in[idx++];
    const float* bo = (const float*)d_in[idx++];

    // Workspace carving (256B aligned). Total ~24 MB.
    char* wsp = (char*)d_ws;
    auto carve = [&wsp](size_t bytes) -> void* {
        void* p = (void*)wsp;
        wsp += (bytes + 255) & ~(size_t)255;
        return p;
    };
    __bf16* xbf   = (__bf16*)carve((size_t)GB * GS * GI * 2);  // 16 MB
    __bf16* w0gx  = (__bf16*)carve((size_t)GH * GI * 2);
    __bf16* w0gh  = (__bf16*)carve((size_t)GH * GH * 2);
    __bf16* w1gx  = (__bf16*)carve((size_t)GH * GH * 2);
    __bf16* w1gh  = (__bf16*)carve((size_t)GH * GH * 2);
    __bf16* wobf  = (__bf16*)carve((size_t)GO * GH * 2);
    float*  hSt   = (float*)carve((size_t)GL * GB * GH * 4);
    __bf16* hbf   = (__bf16*)carve((size_t)GL * GB * GH * 2);
    __bf16* rhbf  = (__bf16*)carve((size_t)GB * GH * 2);
    float*  zbuf  = (float*)carve((size_t)GB * 4);

    // One-time (per launch) conversions: fp32 -> bf16, weights stay [N][K].
    f32_to_bf16_kernel<<<2048, 256, 0, stream>>>(x, xbf, (long)GB * GS * GI);
    f32_to_bf16_kernel<<<256, 256, 0, stream>>>(lp[0].Wgx, w0gx, (long)GH * GI);
    f32_to_bf16_kernel<<<1024, 256, 0, stream>>>(lp[0].Wgh, w0gh, (long)GH * GH);
    f32_to_bf16_kernel<<<1024, 256, 0, stream>>>(lp[1].Wgx, w1gx, (long)GH * GH);
    f32_to_bf16_kernel<<<1024, 256, 0, stream>>>(lp[1].Wgh, w1gh, (long)GH * GH);
    f32_to_bf16_kernel<<<256, 256, 0, stream>>>(Wo, wobf, (long)GO * GH);

    const long nh = (long)GL * GB * GH;
    zero_h_kernel<<<(int)((nh + 255) / 256), 256, 0, stream>>>(hSt, hbf, nh);

    float* outY = (float*)d_out;
    float* outH = (float*)d_out + (size_t)GB * GS * GO;

    float*  h0   = hSt;
    float*  h1   = hSt + (size_t)GB * GH;
    __bf16* hbf0 = hbf;
    __bf16* hbf1 = hbf + (size_t)GB * GH;

    for (int t = 0; t < GS; ++t) {
        // ---- layer 0: inp = x[:, t, :] (row stride S*I) ----
        gru_gates_kernel<<<GB, 256, 0, stream>>>(
            x + (long)t * GI, (long)GS * GI, GI, h0,
            lp[0].Wzx, lp[0].bz, lp[0].Wzh, lp[0].Wrx, lp[0].br, lp[0].Wrh,
            zbuf, rhbf);
        gemm_wmma_kernel<<<GH / 64, 256, 0, stream>>>(
            xbf + (long)t * GI, GS * GI, GI, w0gx,
            rhbf, GH, GH, w0gh,
            lp[0].bg, GH, /*mode=*/0, zbuf, h0, hbf0, nullptr, 0);

        // ---- layer 1: inp = h0 (new) ----
        gru_gates_kernel<<<GB, 256, 0, stream>>>(
            h0, GH, GH, h1,
            lp[1].Wzx, lp[1].bz, lp[1].Wzh, lp[1].Wrx, lp[1].br, lp[1].Wrh,
            zbuf, rhbf);
        gemm_wmma_kernel<<<GH / 64, 256, 0, stream>>>(
            hbf0, GH, GH, w1gx,
            rhbf, GH, GH, w1gh,
            lp[1].bg, GH, /*mode=*/0, zbuf, h1, hbf1, nullptr, 0);

        // ---- output projection: y[:, t, :] = h1 @ Wo.T + bo ----
        gemm_wmma_kernel<<<GO / 64, 256, 0, stream>>>(
            hbf1, GH, GH, wobf,
            nullptr, 0, 0, nullptr,
            bo, GO, /*mode=*/1, nullptr, nullptr, nullptr,
            outY + (long)t * GO, (long)GS * GO);
    }

    copy_hidden_kernel<<<(int)((nh + 255) / 256), 256, 0, stream>>>(hSt, outH);
}
